// DEEP_GATGNN_66090956751316
// MI455X (gfx1250) — matrix operand order repaired and checked by reading, not compiled, not run on hardware
//
#include <hip/hip_runtime.h>
#include <hip/hip_bf16.h>

// ---------------------------------------------------------------------------
// DEEP_GATGNN on MI455X (gfx1250): bf16 WMMA GEMM + fused attention dot,
// then BN / segment-softmax / segment-sum via atomics.
// ---------------------------------------------------------------------------

typedef __attribute__((ext_vector_type(16))) __bf16 v16bf;
typedef __attribute__((ext_vector_type(8)))  __bf16 v8bf;
typedef __attribute__((ext_vector_type(8)))  float  v8f;

#define A_STRIDE 136   // 128 + 8 bf16 pad -> 272B row stride, spreads LDS banks

// Fast softplus: max(x,0) + log(1+exp(-|x|)) using HW transcendentals
// (v_exp_f32 + v_log_f32). Abs error < ~2^-21 — negligible vs bf16 GEMM.
__device__ __forceinline__ float softplusf(float x) {
  return fmaxf(x, 0.0f) + __logf(1.0f + __expf(-fabsf(x)));
}

// -------------------------------- init ------------------------------------
__global__ void gat_init(float* __restrict__ out, const float* __restrict__ bias,
                         unsigned* __restrict__ nodemax, float* __restrict__ nodesum,
                         float* __restrict__ stats, int N) {
  int t = blockIdx.x * blockDim.x + threadIdx.x;
  if (t < N * 64) out[t] = bias[t & 63];
  if (t < N * 4) { nodemax[t] = 0u; nodesum[t] = 0.0f; }
  if (t < 8) stats[t] = 0.0f;
}

// ---------------------- W fp32 -> bf16 fragment order ----------------------
// Wfrag[((ntile*4 + ks)*32 + lane)*16 + t] = bf16(W[K][Ncol])
//   K = ks*32 + ((lane>>4)<<4) + t ;  Ncol = ntile*16 + (lane&15)
__global__ void gat_wconv(const float* __restrict__ W, unsigned short* __restrict__ Wf) {
  int t = blockIdx.x * blockDim.x + threadIdx.x;
  if (t >= 32768) return;
  int el = t & 15;
  int ln = (t >> 4) & 31;
  int ks = (t >> 9) & 3;
  int nt = t >> 11;
  int K  = ks * 32 + ((ln >> 4) << 4) + el;
  int Nc = nt * 16 + (ln & 15);
  union { __bf16 b; unsigned short u; } cv;
  cv.b = (__bf16)W[K * 256 + Nc];
  Wf[t] = cv.u;
}

// ---------------------- GEMM + softplus + attention dot --------------------
// Per 16-edge tile: out_i/out_j = softplus([x_row, edge_attr] @ W) via bf16 WMMA.
// out_i consumed in-kernel for the att dot; out_j stored. alpha1 = softplus(dot).
__global__ void __launch_bounds__(256)
gat_gemm(const float* __restrict__ x, const float* __restrict__ ea,
         const float* __restrict__ att, const unsigned short* __restrict__ Wf,
         const int* __restrict__ eidx,          // [2,E] (int32)
         float* __restrict__ outj,              // [E,256]
         float* __restrict__ alpha1,            // [E,4]
         float* __restrict__ stats,             // [8]: sum[4], sumsq[4]
         int E, int numTiles) {
  extern __shared__ char smem[];
  __bf16* sW  = (__bf16*)smem;                       // 65536 B
  __bf16* sAi = (__bf16*)(smem + 65536);             //  4352 B
  __bf16* sAj = (__bf16*)(smem + 65536 + 4352);      //  4352 B
  float*  sAl = (float*)(smem + 74240);              //   256 B (16 rows x 4 heads)
  float*  sRd = (float*)(smem + 74496);              //  1024 B
  int*    sIx = (int*)(smem + 75520);                //   128 B

  const int tid  = threadIdx.x;
  const int lane = tid & 31;
  const int wave = tid >> 5;

  { // stage W fragments (64 KB) into LDS
    const uint4* src = (const uint4*)Wf;
    uint4* dst = (uint4*)sW;
    for (int p = tid; p < 4096; p += 256) dst[p] = src[p];
  }

  float s_acc = 0.0f, ss_acc = 0.0f;
  const int mrow = tid >> 2, mh = tid & 3;           // finalize mapping (tid<64)

  for (int tile = blockIdx.x; tile < numTiles; tile += gridDim.x) {
    const int e0 = tile * 16;
    __syncthreads();
    if (tid < 32) {
      int r = tid & 15, e = e0 + r, v = 0;
      if (e < E) v = (tid < 16) ? eidx[e] : eidx[E + e];
      sIx[tid] = v;
    }
    if (tid < 64) sAl[tid] = 0.0f;
    __syncthreads();
    // gather + convert A tiles: cols 0..63 = x[row], 64..127 = edge_attr[e]
    for (int p = tid; p < 2048; p += 256) {
      int m = p >> 7, c = p & 127, e = e0 + m;
      float vi, vj;
      if (c < 64) {
        vi = x[(size_t)sIx[m] * 64 + c];
        vj = x[(size_t)sIx[16 + m] * 64 + c];
      } else {
        float v = (e < E) ? ea[(size_t)e * 64 + (c - 64)] : 0.0f;
        vi = v; vj = v;
      }
      sAi[m * A_STRIDE + c] = (__bf16)vi;
      sAj[m * A_STRIDE + c] = (__bf16)vj;
    }
    __syncthreads();
    // 32 tile-jobs = {out_i,out_j} x 16 N-tiles; wave w -> jobs 4w..4w+3
    for (int q = 0; q < 4; ++q) {
      const int job = wave * 4 + q;
      const int mat = job >> 4;                      // 0=out_i, 1=out_j
      const int nt  = job & 15;
      const __bf16* A = mat ? sAj : sAi;
      const int arow = (lane & 15) * A_STRIDE;
      const int koff = (lane >> 4) << 4;
      v8f acc = {0.f, 0.f, 0.f, 0.f, 0.f, 0.f, 0.f, 0.f};
#pragma unroll
      for (int ks = 0; ks < 4; ++ks) {
        int kb = ks * 32 + koff;
        v8bf lo = *(const v8bf*)(A + arow + kb);
        v8bf hi = *(const v8bf*)(A + arow + kb + 8);
        v16bf a = __builtin_shufflevector(lo, hi, 0,1,2,3,4,5,6,7,8,9,10,11,12,13,14,15);
        v16bf b = *(const v16bf*)(sW + (((nt * 4 + ks) * 32 + lane) << 4));
        acc = __builtin_amdgcn_wmma_f32_16x16x32_bf16(false, a, false, b,
                                                      (short)0, acc, false, false);
      }
      const int h = nt >> 2;
      const int d = ((nt & 3) << 4) + (lane & 15);
      const float av = att[h * 128 + (mat ? 64 : 0) + d];
      const int mbase = (lane >> 4) << 3;
#pragma unroll
      for (int v = 0; v < 8; ++v) {
        float o = softplusf(acc[v]);
        int m = mbase + v;
        if (mat && (e0 + m) < E)
          outj[(size_t)(e0 + m) * 256 + nt * 16 + (lane & 15)] = o;
        atomicAdd(&sAl[m * 4 + h], o * av);
      }
    }
    __syncthreads();
    if (tid < 64) {
      int e = e0 + mrow;
      if (e < E) {
        float a1 = softplusf(sAl[tid]);
        alpha1[(size_t)e * 4 + mh] = a1;
        s_acc += a1;
        ss_acc += a1 * a1;
      }
    }
  }
  // BN partial sums: one atomic set per block
  __syncthreads();
  sRd[tid] = (tid < 64) ? s_acc : 0.0f;
  __syncthreads();
  if (tid < 4) {
    float t = 0.0f;
    for (int k = tid; k < 256; k += 4) t += sRd[k];
    atomicAdd(&stats[tid], t);
  }
  __syncthreads();
  sRd[tid] = (tid < 64) ? ss_acc : 0.0f;
  __syncthreads();
  if (tid < 4) {
    float t = 0.0f;
    for (int k = tid; k < 256; k += 4) t += sRd[k];
    atomicAdd(&stats[4 + tid], t);
  }
}

// ------------------- BN + softplus + per-node segment max ------------------
__global__ void gat_bnmax(const float* __restrict__ alpha1, float* __restrict__ alpha2,
                          const float* __restrict__ stats, const float* __restrict__ gam,
                          const float* __restrict__ bet, const int* __restrict__ eidx,
                          unsigned* __restrict__ nodemax, int E) {
  int t = blockIdx.x * blockDim.x + threadIdx.x;
  if (t >= E * 4) return;
  int e = t >> 2, h = t & 3;
  float inv = 1.0f / (float)E;
  float mu  = stats[h] * inv;
  float var = stats[4 + h] * inv - mu * mu;
  float a = (alpha1[t] - mu) * rsqrtf(var + 1e-5f) * gam[h] + bet[h];
  a = softplusf(a);                                  // > 0, so uint max is valid
  alpha2[t] = a;
  atomicMax(&nodemax[eidx[e] * 4 + h], __float_as_uint(a));
}

// ------------------------- exp(a - max) + segment sum ----------------------
__global__ void gat_expsum(float* __restrict__ alpha2, const unsigned* __restrict__ nodemax,
                           float* __restrict__ nodesum, const int* __restrict__ eidx, int E) {
  int t = blockIdx.x * blockDim.x + threadIdx.x;
  if (t >= E * 4) return;
  int e = t >> 2, h = t & 3;
  int i = eidx[e];
  float ex = __expf(alpha2[t] - __uint_as_float(nodemax[i * 4 + h]));
  alpha2[t] = ex;
  atomicAdd(&nodesum[i * 4 + h], ex);
}

// ------------------- weighted message aggregation (+ head mean) ------------
__global__ void gat_aggr(const float* __restrict__ outj, const float* __restrict__ alpha2,
                         const float* __restrict__ nodesum, const int* __restrict__ eidx,
                         float* __restrict__ out, int E) {
  long long t = (long long)blockIdx.x * blockDim.x + threadIdx.x;
  if (t >= (long long)E * 64) return;
  int e = (int)(t >> 6), d = (int)(t & 63);
  int i = eidx[e];
  float acc = 0.0f;
#pragma unroll
  for (int h = 0; h < 4; ++h) {
    float w = alpha2[e * 4 + h] / (nodesum[i * 4 + h] + 1e-16f);
    acc += outj[(size_t)e * 256 + h * 64 + d] * w;
  }
  atomicAdd(&out[(size_t)i * 64 + d], 0.25f * acc);
}

// ---------------------------------------------------------------------------
extern "C" void kernel_launch(void* const* d_in, const int* in_sizes, int n_in,
                              void* d_out, int out_size, void* d_ws, size_t ws_size,
                              hipStream_t stream) {
  const float* x    = (const float*)d_in[0];
  const float* ea   = (const float*)d_in[1];
  const float* W    = (const float*)d_in[2];
  const float* att  = (const float*)d_in[3];
  const float* bias = (const float*)d_in[4];
  const float* gam  = (const float*)d_in[5];
  const float* bet  = (const float*)d_in[6];
  const int*   eidx = (const int*)d_in[7];   // [2,E]
  float* out = (float*)d_out;

  const int N = in_sizes[0] / 64;
  const int E = in_sizes[1] / 64;
  const int numTiles = (E + 15) / 16;

  // workspace carving (256B aligned)
  auto al = [](size_t v) { return (v + 255) & ~(size_t)255; };
  char* p = (char*)d_ws;
  size_t off = 0;
  unsigned short* Wf = (unsigned short*)(p + off); off = al(off + 32768 * 2);
  float* outj    = (float*)(p + off); off = al(off + (size_t)E * 256 * 4);
  float* alpha1  = (float*)(p + off); off = al(off + (size_t)E * 4 * 4);
  float* alpha2  = (float*)(p + off); off = al(off + (size_t)E * 4 * 4);
  float* stats   = (float*)(p + off); off = al(off + 8 * 4);
  unsigned* nmax = (unsigned*)(p + off); off = al(off + (size_t)N * 4 * 4);
  float* nsum    = (float*)(p + off); off = al(off + (size_t)N * 4 * 4);
  (void)ws_size; (void)n_in; (void)out_size;

  const int initN = N * 64;
  gat_init<<<(initN + 255) / 256, 256, 0, stream>>>(out, bias, nmax, nsum, stats, N);
  gat_wconv<<<128, 256, 0, stream>>>(W, Wf);

  int blocks = numTiles < 2048 ? numTiles : 2048;
  const size_t smem = 75648;   // 64K Wfrag + 2x A tile + alpha/reduce/idx
  gat_gemm<<<blocks, 256, smem, stream>>>(x, ea, att, Wf, eidx, outj, alpha1,
                                          stats, E, numTiles);

  gat_bnmax<<<(E * 4 + 255) / 256, 256, 0, stream>>>(alpha1, alpha2, stats, gam, bet,
                                                     eidx, nmax, E);
  gat_expsum<<<(E * 4 + 255) / 256, 256, 0, stream>>>(alpha2, nmax, nsum, eidx, E);

  long long aggN = (long long)E * 64;
  gat_aggr<<<(int)((aggN + 255) / 256), 256, 0, stream>>>(outj, alpha2, nsum, eidx, out, E);
}
